// NetworkLayer_23115513987269
// MI455X (gfx1250) — matrix operand order repaired and checked by reading, not compiled, not run on hardware
//
#include <hip/hip_runtime.h>

// MI455X / gfx1250, wave32. bf16 WMMA (16x16x32, f32 accum) for all GEMMs.
// h1 kernel stages its B operand through LDS with the Tensor Data Mover
// (tensor_load_to_lds, double buffered, s_wait_tensorcnt) and generates the
// sqrt(x_i . x_j) A fragments in registers with raw v_sqrt_f32.
//
// Workspace layout (requires ~50 MB of d_ws):
//   w1dT bf16 [256][2048]   @ 0          (1,048,576 B)
//   w2T  bf16 [256][256]    @ 1,048,576  (  131,072 B)
//   w3T  bf16 [128][256]    @ 1,179,648  (   65,536 B)
//   h1   bf16 [16*2048][256]@ 1,245,184  (16,777,216 B)
//   h2   bf16 [16*2048][256]@ 18,022,400 (16,777,216 B)
//   fk   f32  [16*2048][128]@ 34,799,616 (16,777,216 B)

typedef __bf16 bf16_t;
typedef __attribute__((ext_vector_type(16))) __bf16 v16bf;
typedef __attribute__((ext_vector_type(8)))  __bf16 v8bf;
typedef __attribute__((ext_vector_type(8)))  float  v8f;
typedef unsigned int       u32;
typedef unsigned long long u64;
typedef __attribute__((ext_vector_type(4))) u32 v4u;
typedef __attribute__((ext_vector_type(8))) u32 v8u;

constexpr int Bc    = 16;
constexpr int Nc    = 2048;
constexpr int Hc    = 256;
constexpr int KOUTc = 128;
constexpr float NEG_SLOPE = 0.01f;

// ---------------------------------------------------------------------------
// TDM: 2D tile load Global -> LDS via tensor_load_to_lds (2 descriptor groups)
// D# bit layout per CDNA5 ISA ch.8 (group0 128b, group1 256b).
// All arguments must be wave-uniform.
// ---------------------------------------------------------------------------
__device__ __forceinline__ void tdm_load_tile_2d(
    u32 lds_byte_off, const void* gptr,
    u32 tile_d0, u32 tile_d1,          // tile elems: d0 contiguous, d1 rows
    u32 tensor_d0, u32 tensor_d1,      // tensor extents (OOB check)
    u32 stride_d0_elems)               // row stride in elements
{
    u64 ga = (u64)(uintptr_t)gptr;
    v4u g0;
    g0.x = 0x1u;                                   // count=1, user descriptor
    g0.y = lds_byte_off;                           // lds_addr
    g0.z = (u32)ga;                                // global_addr[31:0]
    g0.w = ((u32)(ga >> 32) & 0x01FFFFFFu)         // global_addr[56:32]
           | (2u << 30);                           // type = 2 ("image")
    v8u g1;
    g1.s0 = (1u << 16);                            // data_size=1 (2B), mask=0
    g1.s1 = (tensor_d0 & 0xFFFFu) << 16;           // tensor_dim0[15:0]
    g1.s2 = (tensor_d0 >> 16) | ((tensor_d1 & 0xFFFFu) << 16); // d0 hi | d1 lo
    g1.s3 = (tensor_d1 >> 16) | (tile_d0 << 16);   // d1 hi | tile_dim0
    g1.s4 = tile_d1;                               // tile_dim1 (tile_dim2 = 0)
    g1.s5 = stride_d0_elems;                       // tensor_dim0_stride lo
    g1.s6 = 0u;                                    // stride hi | dim1_stride lo
    g1.s7 = 0u;
    asm volatile("tensor_load_to_lds %0, %1" :: "s"(g0), "s"(g1) : "memory");
}

// ---------------------------------------------------------------------------
// Kernel 0: convert + transpose weights to bf16 (row = output column).
// ---------------------------------------------------------------------------
__global__ __launch_bounds__(256) void prep_kernel(
    const float* __restrict__ W1, const float* __restrict__ W2,
    const float* __restrict__ W3, bf16_t* __restrict__ w1dT,
    bf16_t* __restrict__ w2T, bf16_t* __restrict__ w3T)
{
    int i = blockIdx.x * 256 + threadIdx.x;
    if (i < Hc * Nc) {            // w1dT[n][k] = W1[7+k][n]
        int n = i >> 11, k = i & 2047;
        w1dT[i] = (bf16_t)W1[(size_t)(7 + k) * Hc + n];
    }
    if (i < Hc * Hc) {            // w2T[n][k] = W2[k][n]
        int n = i >> 8, k = i & 255;
        w2T[i] = (bf16_t)W2[(size_t)k * Hc + n];
    }
    if (i < KOUTc * Hc) {         // w3T[n][k] = W3[k][n]
        int n = i >> 8, k = i & 255;
        w3T[i] = (bf16_t)W3[(size_t)k * KOUTc + n];
    }
}

// ---------------------------------------------------------------------------
// Kernel 1: fused  h1 = LeakyReLU([u, norm, proj, sqrt(X X^T)] @ W1 + b1)
// One workgroup = one (batch, 128-row tile): 128x256 output, K = 2048.
// A fragments (dots) generated in registers from x staged in LDS; B tiles
// DMA'd into double-buffered LDS by the Tensor Data Mover (1 slice / wave).
// ---------------------------------------------------------------------------
__global__ __launch_bounds__(256) void h1_kernel(
    const float* __restrict__ x, const float* __restrict__ u,
    const float* __restrict__ basis, const float* __restrict__ W1,
    const float* __restrict__ b1v, const bf16_t* __restrict__ w1dT,
    bf16_t* __restrict__ h1)
{
    __shared__ float  xs[Nc * 3];       // all x for this batch (24 KB)
    __shared__ float  scal[128][8];     // u0,u1,norm,proj0..3 per tile row
    __shared__ float  w1s[7 * Hc];      // small-feature weight rows
    __shared__ float  b1s[Hc];
    __shared__ bf16_t ldsB[2][Hc * 32]; // double-buffered B tile (2 x 16 KB)

    const int b   = blockIdx.x >> 4;    // 16 row tiles per batch
    const int rt  = blockIdx.x & 15;
    const int tid = threadIdx.x;

    const float* xb = x + (size_t)b * Nc * 3;
    for (int i = tid; i < Nc * 3; i += 256) xs[i] = xb[i];
    for (int i = tid; i < 7 * Hc; i += 256) w1s[i] = W1[i];
    b1s[tid] = b1v[tid];
    __syncthreads();

    if (tid < 128) {                    // per-row small scalars
        int r = rt * 128 + tid;
        float x0 = xs[r * 3 + 0], x1 = xs[r * 3 + 1], x2 = xs[r * 3 + 2];
        scal[tid][0] = u[b * 2 + 0];
        scal[tid][1] = u[b * 2 + 1];
        scal[tid][2] = __builtin_amdgcn_sqrtf(x0 * x0 + x1 * x1 + x2 * x2);
#pragma unroll
        for (int k = 0; k < 4; ++k) {
            const float* bs = basis + ((size_t)b * 4 + k) * 3;
            scal[tid][3 + k] = x0 * bs[0] + x1 * bs[1] + x2 * bs[2];
        }
    }

    const int wave = tid >> 5, lane = tid & 31;
    const int c = lane >> 4, ln = lane & 15;
    const int wrow = wave >> 1;         // 4 wave-rows x 2 wave-cols
    const int wcol = wave & 1;
    const int rowbase = rt * 128 + wrow * 32;   // row within batch
    const int colbase = wcol * 128;

    // wave-uniform slice id for the TDM descriptors
    const int wv = __builtin_amdgcn_readfirstlane(wave);

    // TDM prologue: each wave DMAs rows [wv*32, wv*32+32) of B tile j=0.
    tdm_load_tile_2d((u32)(uintptr_t)&ldsB[0][wv * 32 * 32],
                     w1dT + (size_t)(wv * 32) * Nc,
                     /*tile*/ 32, 32, /*tensor*/ Nc, Hc, /*stride*/ Nc);

    __syncthreads();                    // xs/scal ready (TDM overlaps)

    // this lane's x rows for the two 16-row A strips (fixed across K loop)
    float xr[2][3];
#pragma unroll
    for (int s = 0; s < 2; ++s) {
        int r = rowbase + s * 16 + ln;
        xr[s][0] = xs[r * 3 + 0];
        xr[s][1] = xs[r * 3 + 1];
        xr[s][2] = xs[r * 3 + 2];
    }

    const v8f vzero = {0.f, 0.f, 0.f, 0.f, 0.f, 0.f, 0.f, 0.f};
    v8f acc[2][8];
#pragma unroll
    for (int s = 0; s < 2; ++s)
#pragma unroll
        for (int t = 0; t < 8; ++t) acc[s][t] = vzero;

    int p = 0;
    for (int j = 0; j < Nc; j += 32, p ^= 1) {
        // issue DMA for the next K-step tile (wraps on last iter so there are
        // always 2 outstanding and s_wait_tensorcnt(1) is exact)
        int jn = (j + 32) & (Nc - 1);
        tdm_load_tile_2d((u32)(uintptr_t)&ldsB[p ^ 1][wv * 32 * 32],
                         w1dT + (size_t)(wv * 32) * Nc + jn,
                         32, 32, Nc, Hc, Nc);

        // A fragments (overlap with DMA): A[m][k] = sqrt(x_m . x_k), bf16,
        // contiguous 16-bit 16x32 layout: k = 16*c + e
        v16bf af[2];
#pragma unroll
        for (int s = 0; s < 2; ++s) {
            union { v16bf v; bf16_t e[16]; } a;
#pragma unroll
            for (int e2 = 0; e2 < 16; ++e2) {
                int k = j + e2 + 16 * c;
                float d = xr[s][0] * xs[k * 3 + 0] +
                          xr[s][1] * xs[k * 3 + 1] +
                          xr[s][2] * xs[k * 3 + 2];
                a.e[e2] = (bf16_t)__builtin_amdgcn_sqrtf(d);
            }
            af[s] = a.v;
        }

        __builtin_amdgcn_s_wait_tensorcnt(1);  // our slice of buffer p done
        __syncthreads();                       // all slices of buffer p done

        // B fragments from LDS tile, 16 WMMAs
#pragma unroll
        for (int t = 0; t < 8; ++t) {
            const bf16_t* bp =
                &ldsB[p][(colbase + t * 16 + ln) * 32 + 16 * c];
            v16bf bf = *(const v16bf*)bp;
            acc[0][t] = __builtin_amdgcn_wmma_f32_16x16x32_bf16(
                false, af[0], false, bf, (short)0, acc[0][t], false, false);
            acc[1][t] = __builtin_amdgcn_wmma_f32_16x16x32_bf16(
                false, af[1], false, bf, (short)0, acc[1][t], false, false);
        }
        __syncthreads();               // reads done before buffer p is re-DMA'd
    }
    __builtin_amdgcn_s_wait_tensorcnt(0);      // drain trailing wrap-around DMA

    // epilogue: small features + bias + LeakyReLU, store bf16
#pragma unroll
    for (int s = 0; s < 2; ++s)
#pragma unroll
        for (int t = 0; t < 8; ++t) {
            int col = colbase + t * 16 + ln;
#pragma unroll
            for (int e = 0; e < 8; ++e) {
                int mrow = wrow * 32 + s * 16 + 8 * c + e; // row in 128-tile
                float v = acc[s][t][e] + b1s[col];
#pragma unroll
                for (int f = 0; f < 7; ++f)
                    v += scal[mrow][f] * w1s[f * Hc + col];
                v = v > 0.f ? v : v * NEG_SLOPE;
                int grow = rt * 128 + mrow;
                h1[((size_t)b * Nc + grow) * Hc + col] = (bf16_t)v;
            }
        }
}

// ---------------------------------------------------------------------------
// Kernels 2/3: generic bf16 WMMA GEMM, A[M][K] bf16, Bt[N][K] bf16 (B^T).
// 128-row block tile, 8 waves. SUBS = 16-row strips per wave (2 -> 32x128
// wave tile for N=256; 1 -> 16x128 wave tile for N=128).
// ---------------------------------------------------------------------------
template <int SUBS, bool LEAKY, bool OUT_BF16>
__global__ __launch_bounds__(256) void gemm_kernel(
    const bf16_t* __restrict__ A, const bf16_t* __restrict__ Bt,
    const float* __restrict__ bias, void* __restrict__ Out,
    int K, int Ncols)
{
    constexpr int WROWS = 128 / (SUBS * 16);    // waves along M
    const int tid = threadIdx.x;
    const int wave = tid >> 5, lane = tid & 31;
    const int c = lane >> 4, ln = lane & 15;
    const int wrow = wave % WROWS;
    const int wcol = wave / WROWS;
    const int rowbase = blockIdx.x * 128 + wrow * (SUBS * 16);
    const int colbase = wcol * 128;

    const v8f vzero = {0.f, 0.f, 0.f, 0.f, 0.f, 0.f, 0.f, 0.f};
    v8f acc[SUBS][8];
#pragma unroll
    for (int s = 0; s < SUBS; ++s)
#pragma unroll
        for (int t = 0; t < 8; ++t) acc[s][t] = vzero;

    for (int j = 0; j < K; j += 32) {
        v16bf af[SUBS];
#pragma unroll
        for (int s = 0; s < SUBS; ++s) {
            union { v16bf v; v8bf h[2]; } a;
            const bf16_t* ap =
                A + (size_t)(rowbase + s * 16 + ln) * K + j + 16 * c;
            a.h[0] = *(const v8bf*)ap;          // k = j + 16c + 0..7
            a.h[1] = *(const v8bf*)(ap + 8);    // k = j + 16c + 8..15
            af[s] = a.v;
        }
#pragma unroll
        for (int t = 0; t < 8; ++t) {
            const bf16_t* bp =
                Bt + (size_t)(colbase + t * 16 + ln) * K + j + 16 * c;
            v16bf bf = *(const v16bf*)bp;
            if (j + 32 < K) __builtin_prefetch(bp + 32, 0, 3);
#pragma unroll
            for (int s = 0; s < SUBS; ++s)
                acc[s][t] = __builtin_amdgcn_wmma_f32_16x16x32_bf16(
                    false, af[s], false, bf, (short)0, acc[s][t], false, false);
        }
    }

#pragma unroll
    for (int s = 0; s < SUBS; ++s)
#pragma unroll
        for (int t = 0; t < 8; ++t) {
            int col = colbase + t * 16 + ln;
#pragma unroll
            for (int e = 0; e < 8; ++e) {
                int grow = rowbase + s * 16 + 8 * c + e;
                float v = acc[s][t][e] + bias[col];
                if (LEAKY) v = v > 0.f ? v : v * NEG_SLOPE;
                size_t idx = (size_t)grow * Ncols + col;
                if (OUT_BF16) ((bf16_t*)Out)[idx] = (bf16_t)v;
                else          ((float*)Out)[idx]  = v;
            }
        }
}

// ---------------------------------------------------------------------------
// Kernel 4: out[b][o][d] = (1/N) * sum_n fk[b][n][o] * x[b][n][d]
// ---------------------------------------------------------------------------
__global__ __launch_bounds__(256) void reduce_kernel(
    const float* __restrict__ fk, const float* __restrict__ x,
    float* __restrict__ out)
{
    __shared__ float sm[3][256];
    const int b = blockIdx.x >> 7;      // 128 output channels per batch
    const int o = blockIdx.x & 127;
    const int t = threadIdx.x;

    float a0 = 0.f, a1 = 0.f, a2 = 0.f;
    for (int n = t; n < Nc; n += 256) {
        float f = fk[((size_t)b * Nc + n) * KOUTc + o];
        const float* xp = x + ((size_t)b * Nc + n) * 3;
        a0 += f * xp[0];
        a1 += f * xp[1];
        a2 += f * xp[2];
    }
    sm[0][t] = a0; sm[1][t] = a1; sm[2][t] = a2;
    __syncthreads();
    for (int st = 128; st > 0; st >>= 1) {
        if (t < st) {
            sm[0][t] += sm[0][t + st];
            sm[1][t] += sm[1][t + st];
            sm[2][t] += sm[2][t + st];
        }
        __syncthreads();
    }
    if (t == 0) {
        const float inv = 1.0f / (float)Nc;
        float* op = out + ((size_t)b * KOUTc + o) * 3;
        op[0] = sm[0][0] * inv;
        op[1] = sm[1][0] * inv;
        op[2] = sm[2][0] * inv;
    }
}

// ---------------------------------------------------------------------------
extern "C" void kernel_launch(void* const* d_in, const int* in_sizes, int n_in,
                              void* d_out, int out_size, void* d_ws,
                              size_t ws_size, hipStream_t stream)
{
    (void)in_sizes; (void)n_in; (void)out_size; (void)ws_size;
    const float* x     = (const float*)d_in[0];
    const float* u     = (const float*)d_in[1];
    const float* basis = (const float*)d_in[2];
    const float* W1    = (const float*)d_in[3];
    const float* b1    = (const float*)d_in[4];
    const float* W2    = (const float*)d_in[5];
    const float* b2    = (const float*)d_in[6];
    const float* W3    = (const float*)d_in[7];
    const float* b3    = (const float*)d_in[8];

    char* ws = (char*)d_ws;
    bf16_t* w1dT = (bf16_t*)(ws);
    bf16_t* w2T  = (bf16_t*)(ws + 1048576);
    bf16_t* w3T  = (bf16_t*)(ws + 1048576 + 131072);
    bf16_t* h1   = (bf16_t*)(ws + 1245184);
    bf16_t* h2   = (bf16_t*)(ws + 1245184 + 16777216);
    float*  fk   = (float*)(ws + 1245184 + 2 * 16777216);

    prep_kernel<<<2048, 256, 0, stream>>>(W1, W2, W3, w1dT, w2T, w3T);

    // h1: 16 batches x 16 row-tiles (TDM-staged B, register A, 16 WMMA/step)
    h1_kernel<<<Bc * 16, 256, 0, stream>>>(x, u, basis, W1, b1, w1dT, h1);

    // h2 = LeakyReLU(h1 @ W2 + b2):  M=32768, K=256, N=256
    gemm_kernel<2, true, true><<<(Bc * Nc) / 128, 256, 0, stream>>>(
        h1, w2T, b2, (void*)h2, Hc, Hc);

    // fk = h2 @ W3 + b3:  M=32768, K=256, N=128, f32 out
    gemm_kernel<1, false, false><<<(Bc * Nc) / 128, 256, 0, stream>>>(
        h2, w3T, b3, (void*)fk, Hc, KOUTc);

    // out = einsum('bio,bid->bod', fk, x) / N
    reduce_kernel<<<Bc * KOUTc, 256, 0, stream>>>(fk, x, (float*)d_out);
}
